// EquivariantUpdate_35570919145942
// MI455X (gfx1250) — matrix operand order repaired and checked by reading, not compiled, not run on hardware
//
#include <hip/hip_runtime.h>
#include <hip/hip_bf16.h>

typedef __attribute__((ext_vector_type(16))) __bf16 v16bf;
typedef __attribute__((ext_vector_type(8)))  float  v8f;

#define HIDDEN   128
#define IN_DIM   257
#define KPAD     288            // 9 chunks of 32 (257 padded)
#define TILE_M   32             // edges per block-tile (two 16-row WMMA tiles)
#define NCHUNK   (TILE_M * (KPAD / 8))   // staging chunks of 8 elements
#define NORM_INV (1.0f / 100.0f)

__device__ __forceinline__ float silu_f(float x) {
    // x * sigmoid(x); v_rcp_f32 instead of IEEE division microcode
    return x * __builtin_amdgcn_rcpf(1.0f + __expf(-x));
}

__device__ __forceinline__ __bf16 f32_to_bf16_rn(float x) {
    unsigned u = __builtin_bit_cast(unsigned, x) + 0x8000u;  // round-half-up
    unsigned short s = (unsigned short)(u >> 16);
    return __builtin_bit_cast(__bf16, s);
}

// Pack hi16(f0) | hi16(f1)<<16 : one v_perm_b32 (bf16 truncation of two floats)
__device__ __forceinline__ unsigned pack_bf16x2(float f0, float f1) {
    return __builtin_amdgcn_perm(__builtin_bit_cast(unsigned, f1),
                                 __builtin_bit_cast(unsigned, f0), 0x07060302u);
}

// Same but with round-half-up before truncation (used for weights).
__device__ __forceinline__ unsigned pack_bf16x2_rn(float f0, float f1) {
    const unsigned u0 = __builtin_bit_cast(unsigned, f0) + 0x8000u;
    const unsigned u1 = __builtin_bit_cast(unsigned, f1) + 0x8000u;
    return __builtin_amdgcn_perm(u1, u0, 0x07060302u);
}

// A fragment (16x32 bf16, M=edges, K=features) from an LDS tile, row-major [rows][ld] elems.
// Wave32 layout: lane L -> M = L%16, g = L/16; element e=2v+h -> K = 16*(v>=4) + 8*g + 2*(v&3) + h.
__device__ __forceinline__ v16bf load_a_frag(const __bf16* base, int ld, int m, int g, int col0) {
    v16bf a;
#pragma unroll
    for (int v = 0; v < 8; ++v) {
        const int K = ((v & 4) ? 16 : 0) + 8 * g + 2 * (v & 3);
        const __bf16* p = base + m * ld + col0 + K;   // dword-aligned pair
        a[2 * v]     = p[0];
        a[2 * v + 1] = p[1];
    }
    return a;
}

// B fragment (32x16 bf16, K=features, N=hidden cols) from global f32 weights
// W[rows][128] row-major. Wave32 layout: lane L -> N = L%16, g = L/16; element e=2p+h -> K = 16*(p>=4) + 8*g + 2*(p&3) + h.
// Rows >= `rows` (the K-padding region) are zero. Pairs of rows pack via v_perm.
__device__ __forceinline__ v16bf load_b_frag(const float* __restrict__ W, int rows,
                                             int row0, int g, int col) {
    union { unsigned u[8]; v16bf v; } b;
#pragma unroll
    for (int p = 0; p < 8; ++p) {
        const int r0 = row0 + ((p & 4) ? 16 : 0) + 8 * g + 2 * (p & 3);
        const float f0 = (r0     < rows) ? W[r0 * HIDDEN + col]       : 0.0f;
        const float f1 = (r0 + 1 < rows) ? W[(r0 + 1) * HIDDEN + col] : 0.0f;
        b.u[p] = pack_bf16x2_rn(f0, f1);
    }
    return b.v;
}

__global__ __launch_bounds__(256) void egnn_edge_mlp_kernel(
    const float* __restrict__ h, const int* __restrict__ edge_index,
    const float* __restrict__ coord_diff, const float* __restrict__ edge_attr,
    const float* __restrict__ edge_mask,
    const float* __restrict__ W1, const float* __restrict__ b1,
    const float* __restrict__ W2, const float* __restrict__ b2,
    const float* __restrict__ W3,
    float* __restrict__ agg, int E, int ntiles)
{
    __shared__ __bf16 s_inp[TILE_M * KPAD];    // 18 KB: [h[row] | h[col] | e_attr | 0]
    __shared__ __bf16 s_x1[TILE_M * HIDDEN];   //  8 KB: layer-1 activations (exchange)
    __shared__ float  s_scale[TILE_M];         // per-edge scalar from layer 3

    const int tid  = threadIdx.x;
    const int wave = tid >> 5;      // 8 waves; wave w owns output columns 16w..16w+15
    const int lane = tid & 31;
    const int g    = lane >> 4;
    const int nl   = lane & 15;
    const int col  = wave * 16 + nl;     // hidden column owned by this lane

    // ---- Register-resident weight fragments (amortized over all tiles) ----
    v16bf w1f[9];
#pragma unroll
    for (int kc = 0; kc < 9; ++kc) w1f[kc] = load_b_frag(W1, IN_DIM, 32 * kc, g, col);
    v16bf w2f[4];
#pragma unroll
    for (int kc = 0; kc < 4; ++kc) w2f[kc] = load_b_frag(W2, HIDDEN, 32 * kc, g, col);
    const float b1v = b1[col];
    const float b2v = b2[col];
    const float w3v = W3[col];

    for (int tile = blockIdx.x; tile < ntiles; tile += gridDim.x) {
        const int ebase = tile * TILE_M;

        // ---- Stage 32x288 bf16 tile, 8 features per chunk:
        //      2x global_load_b128 + 4x v_perm + 1x ds_store_b128 ----
        for (int c = tid; c < NCHUNK; c += 256) {
            const int mm = c / (KPAD / 8);
            const int k0 = (c - mm * (KPAD / 8)) * 8;
            const int e  = ebase + mm;
            float4 f0 = {0.f, 0.f, 0.f, 0.f};
            float4 f1 = {0.f, 0.f, 0.f, 0.f};
            if (e < E) {
                if (k0 < 2 * HIDDEN) {
                    // merged h[row]/h[col] gather: select index, no exec-mask split
                    const int which = k0 >> 7;                    // 0 -> row, 1 -> col
                    const int idx   = edge_index[e + which * E];
                    const float* src = h + (size_t)idx * HIDDEN + (k0 & (HIDDEN - 1));
                    f0 = *(const float4*)(src);
                    f1 = *(const float4*)(src + 4);
                } else if (k0 == 2 * HIDDEN) {
                    f0.x = edge_attr[e];
                }
            }
            uint4 pk;
            pk.x = pack_bf16x2(f0.x, f0.y);
            pk.y = pack_bf16x2(f0.z, f0.w);
            pk.z = pack_bf16x2(f1.x, f1.y);
            pk.w = pack_bf16x2(f1.z, f1.w);
            *(uint4*)(&s_inp[mm * KPAD + k0]) = pk;   // 16B-aligned (576B row stride)
        }
        if (tid < TILE_M) s_scale[tid] = 0.0f;

        // ---- Prefetch next tile's gathered rows (overlaps with this tile's GEMMs) ----
        {
            const int tnext = tile + gridDim.x;
            if (wave == 7 && tnext < ntiles) {
                const int e2 = tnext * TILE_M + lane;
                if (e2 < E) {
                    const char* pr = (const char*)(h + (size_t)edge_index[e2] * HIDDEN);
                    const char* pc = (const char*)(h + (size_t)edge_index[E + e2] * HIDDEN);
#pragma unroll
                    for (int o = 0; o < 512; o += 128) {       // 512B row = 4 cachelines
                        __builtin_prefetch(pr + o, 0, 3);      // global_prefetch_b8
                        __builtin_prefetch(pc + o, 0, 3);
                    }
                }
            }
        }
        __syncthreads();

        // ---- Layer 1: x1 = silu(inp @ W1 + b1) : two independent WMMA chains ----
        v8f acc0 = {};
        v8f acc1 = {};
#pragma unroll
        for (int kc = 0; kc < 9; ++kc) {
            v16bf a0 = load_a_frag(s_inp, KPAD, nl,      g, 32 * kc);
            v16bf a1 = load_a_frag(s_inp, KPAD, nl + 16, g, 32 * kc);
            acc0 = __builtin_amdgcn_wmma_f32_16x16x32_bf16(false, a0, false, w1f[kc],
                                                           (short)0, acc0, false, false);
            acc1 = __builtin_amdgcn_wmma_f32_16x16x32_bf16(false, a1, false, w1f[kc],
                                                           (short)0, acc1, false, false);
        }
#pragma unroll
        for (int i = 0; i < 8; ++i) {
            // C/D layout: row M = i + 8*g (+16 for second tile), col N = col
            s_x1[(i + 8 * g) * HIDDEN + col]      = f32_to_bf16_rn(silu_f(acc0[i] + b1v));
            s_x1[(i + 8 * g + 16) * HIDDEN + col] = f32_to_bf16_rn(silu_f(acc1[i] + b1v));
        }
        __syncthreads();

        // ---- Layer 2: x2 = silu(x1 @ W2 + b2) ----
        v8f acc2 = {};
        v8f acc3 = {};
#pragma unroll
        for (int kc = 0; kc < 4; ++kc) {
            v16bf a0 = load_a_frag(s_x1, HIDDEN, nl,      g, 32 * kc);
            v16bf a1 = load_a_frag(s_x1, HIDDEN, nl + 16, g, 32 * kc);
            acc2 = __builtin_amdgcn_wmma_f32_16x16x32_bf16(false, a0, false, w2f[kc],
                                                           (short)0, acc2, false, false);
            acc3 = __builtin_amdgcn_wmma_f32_16x16x32_bf16(false, a1, false, w2f[kc],
                                                           (short)0, acc3, false, false);
        }

        // ---- Layer 3: scale[m] = sum_n x2[m][n] * W3[n] ----
        float s0[8], s1[8];
#pragma unroll
        for (int i = 0; i < 8; ++i) {
            s0[i] = silu_f(acc2[i] + b2v) * w3v;
            s1[i] = silu_f(acc3[i] + b2v) * w3v;
        }
#pragma unroll
        for (int i = 0; i < 8; ++i) {
#pragma unroll
            for (int mask = 1; mask <= 8; mask <<= 1) {
                s0[i] += __shfl_xor(s0[i], mask, 32);   // reduce over 16-lane half (n dim)
                s1[i] += __shfl_xor(s1[i], mask, 32);
            }
        }
        if (nl == 0) {
#pragma unroll
            for (int i = 0; i < 8; ++i) {
                atomicAdd(&s_scale[i + 8 * g],      s0[i]);   // ds_add_f32
                atomicAdd(&s_scale[i + 8 * g + 16], s1[i]);
            }
        }
        __syncthreads();

        // ---- Scatter: agg[row] += coord_diff * scale * edge_mask ----
        if (wave == 0) {
            const int e = ebase + lane;
            if (e < E) {
                const float f = s_scale[lane] * edge_mask[e];
                const int r = edge_index[e];
                atomicAdd(&agg[r * 3 + 0], coord_diff[e * 3 + 0] * f);
                atomicAdd(&agg[r * 3 + 1], coord_diff[e * 3 + 1] * f);
                atomicAdd(&agg[r * 3 + 2], coord_diff[e * 3 + 2] * f);
            }
        }
        __syncthreads();   // protect s_inp / s_scale for next tile
    }
}

__global__ void egnn_zero_kernel(float* __restrict__ p, int n) {
    const int i = blockIdx.x * blockDim.x + threadIdx.x;
    if (i < n) p[i] = 0.0f;
}

__global__ void egnn_finalize_kernel(const float* __restrict__ coord,
                                     const float* __restrict__ agg,
                                     const float* __restrict__ node_mask,
                                     float* __restrict__ out, int n3) {
    const int i = blockIdx.x * blockDim.x + threadIdx.x;
    if (i < n3) out[i] = (coord[i] + agg[i] * NORM_INV) * node_mask[i / 3];
}

extern "C" void kernel_launch(void* const* d_in, const int* in_sizes, int n_in,
                              void* d_out, int out_size, void* d_ws, size_t ws_size,
                              hipStream_t stream) {
    const float* h     = (const float*)d_in[0];
    const float* coord = (const float*)d_in[1];
    const int*   eidx  = (const int*)d_in[2];
    const float* cdiff = (const float*)d_in[3];
    const float* eattr = (const float*)d_in[4];
    const float* nmask = (const float*)d_in[5];
    const float* emask = (const float*)d_in[6];
    const float* W1    = (const float*)d_in[7];
    const float* b1    = (const float*)d_in[8];
    const float* W2    = (const float*)d_in[9];
    const float* b2    = (const float*)d_in[10];
    const float* W3    = (const float*)d_in[11];
    float* out = (float*)d_out;
    float* agg = (float*)d_ws;            // N*3 f32 accumulator

    const int N      = in_sizes[1] / 3;
    const int E      = in_sizes[4];
    const int n3     = N * 3;
    const int ntiles = (E + TILE_M - 1) / TILE_M;

    egnn_zero_kernel<<<(n3 + 255) / 256, 256, 0, stream>>>(agg, n3);

    const int nblocks = ntiles < 1024 ? ntiles : 1024;
    egnn_edge_mlp_kernel<<<nblocks, 256, 0, stream>>>(h, eidx, cdiff, eattr, emask,
                                                      W1, b1, W2, b2, W3, agg, E, ntiles);

    egnn_finalize_kernel<<<(n3 + 255) / 256, 256, 0, stream>>>(coord, agg, nmask, out, n3);
}
